// SLSTMNET_69698729279707
// MI455X (gfx1250) — compile-verified, matrix operand order checked by reading, not run on hardware
//
#include <hip/hip_runtime.h>
#include <hip/hip_bf16.h>

// ---------------------------------------------------------------------------
// SLSTM network for MI455X (gfx1250), wave32 + WMMA bf16.
//
//   K0 prep_weights : permute+pad gate rows (H 50->64, 4H 200->256 in 16
//                     tiles of 16 ordered (jb,gate)), bf16-pack weights.
//   K1 transpose_x  : x[b,i,j] -> xT[b,j,i] (coalesced "columns" stream).
//   K2 input_gemm   : Z = seq @ W_ih'^T + bias for all t. 16 waves/block,
//                     each wave owns one gate tile with W-fragments RESIDENT
//                     in registers; one shared A-tile in LDS; 30 consecutive
//                     timesteps per block. A-fragments are bulk-loaded before
//                     the WMMA chain so LDS latency is paid once per task.
//   K3 recurrent    : 256 single-wave blocks (one batch tile each, spread
//                     across WGPs for the latency-bound serial phase).
//                     W_hh' staged to LDS via async global->LDS copy;
//                     per step: 32 recurrent WMMAs + lane-local LSTM math
//                     (gate permutation aligns i/f/g/o per lane) + 2 FC
//                     WMMAs + LIF; Z tile bt+1 prefetched; per-jb fragment loads
//                     hoisted ahead of the WMMA chain.
// ---------------------------------------------------------------------------

typedef __bf16 bf16_t;
typedef __bf16 v16bf __attribute__((ext_vector_type(16)));
typedef __bf16 v8bf  __attribute__((ext_vector_type(8)));
typedef __bf16 v2bf  __attribute__((ext_vector_type(2)));
typedef float  v8f   __attribute__((ext_vector_type(8)));

#define HH    50
#define INF   300
#define NOUT  3
#define BATCH 4096
#define TT    150
#define KP    320   // padded input features (300 -> 320 = 10 * 32)
#define GP    256   // padded gate rows (4*64), 16 tiles of 16
#define HP    64    // padded hidden
#define NBT   256   // batch tiles of 16
#define TPB   30    // timesteps per K2 block (150 = 5 * 30)

__device__ __forceinline__ v16bf mk16(v8bf lo, v8bf hi) {
  v16bf r;
#pragma unroll
  for (int i = 0; i < 8; ++i) { r[i] = lo[i]; r[i + 8] = hi[i]; }
  return r;
}

__device__ __forceinline__ float sigm(float x) {
  return __fdividef(1.f, 1.f + __expf(-x));
}
__device__ __forceinline__ float tanh_fast(float x) {
  x = fminf(fmaxf(x, -15.f), 15.f);
  float e = __expf(2.f * x);
  return __fdividef(e - 1.f, e + 1.f);
}

// --------------------------- K0: weight prep -------------------------------
__global__ void prep_weights(const float* __restrict__ w_ih,
                             const float* __restrict__ w_hh,
                             const float* __restrict__ b_ih,
                             const float* __restrict__ b_hh,
                             const float* __restrict__ fc_w,
                             bf16_t* __restrict__ wbf,   // [GP][KP]
                             bf16_t* __restrict__ whbf,  // [GP][HP]
                             float*  __restrict__ bias,  // [GP]
                             bf16_t* __restrict__ fcbf)  // [16][HP]
{
  int tid = threadIdx.x;  // 256 threads, 1 block
  for (int idx = tid; idx < GP * KP; idx += 256) {
    int n = idx / KP, k = idx % KP;
    int nt = n >> 4, l = n & 15, jb = nt >> 2, g = nt & 3, j = jb * 16 + l;
    float v = (j < HH && k < INF) ? w_ih[(g * HH + j) * INF + k] : 0.f;
    wbf[idx] = (bf16_t)v;
  }
  for (int idx = tid; idx < GP * HP; idx += 256) {
    int n = idx / HP, k = idx % HP;
    int nt = n >> 4, l = n & 15, jb = nt >> 2, g = nt & 3, j = jb * 16 + l;
    float v = (j < HH && k < HH) ? w_hh[(g * HH + j) * HH + k] : 0.f;
    whbf[idx] = (bf16_t)v;
  }
  for (int n = tid; n < GP; n += 256) {
    int nt = n >> 4, l = n & 15, jb = nt >> 2, g = nt & 3, j = jb * 16 + l;
    bias[n] = (j < HH) ? (b_ih[g * HH + j] + b_hh[g * HH + j]) : 0.f;
  }
  for (int idx = tid; idx < 16 * HP; idx += 256) {
    int n = idx / HP, k = idx % HP;
    float v = (n < NOUT && k < HH) ? fc_w[n * HH + k] : 0.f;
    fcbf[idx] = (bf16_t)v;
  }
}

// --------------------------- K1: x transpose -------------------------------
__global__ __launch_bounds__(256) void transpose_x(const float* __restrict__ x,
                                                   float* __restrict__ xT) {
  __shared__ float s[16][17];
  int b = blockIdx.y;
  int ti = blockIdx.x / 10, tj = blockIdx.x % 10;
  int tx = threadIdx.x & 15, ty = threadIdx.x >> 4;
  size_t base = (size_t)b * TT * TT;
  int i = ti * 16 + ty, j = tj * 16 + tx;
  if (i < TT && j < TT) s[ty][tx] = x[base + (size_t)i * TT + j];
  __syncthreads();
  int i2 = ti * 16 + tx, j2 = tj * 16 + ty;
  if (i2 < TT && j2 < TT) xT[base + (size_t)j2 * TT + i2] = s[tx][ty];
}

// --------------------------- K2: batched input GEMM ------------------------
// 16 waves/block; wave w owns gate tile nt=w with W-fragments in registers.
// Block processes one batch tile x 30 consecutive timesteps.
__global__ __launch_bounds__(512) void input_gemm(
    const float* __restrict__ x, const float* __restrict__ xT,
    const bf16_t* __restrict__ wbf, const float* __restrict__ bias,
    float* __restrict__ Z) {
  __shared__ bf16_t atile[16 * KP];  // 10 KB, shared by all 16 waves
  int tid = threadIdx.x, wave = tid >> 5, lane = tid & 31;
  int bt = blockIdx.x & (NBT - 1);
  int tg = blockIdx.x >> 8;  // 0..4
  int b0 = bt * 16;
  int nt = wave;
  int mr = lane & 15, kh = lane >> 4;

  // Resident B-fragments for this wave's gate tile (10 x 8 VGPRs).
  v16bf bfr[10];
  {
    const bf16_t* wrow = wbf + (size_t)(nt * 16 + mr) * KP + kh * 8;
#pragma unroll
    for (int kb = 0; kb < 10; ++kb) {
      const v8bf* q = (const v8bf*)(wrow + kb * 32);
      bfr[kb] = mk16(q[0], q[2]);
    }
  }
  float bv = bias[nt * 16 + mr];

  for (int it = 0; it < TPB; ++it) {
    int t = tg * TPB + it;
    {  // stage A tile: wave w loads batch row m=w (rows + transposed cols)
      int m = wave;
      const float* xr = x + ((size_t)(b0 + m) * TT + t) * TT;   // 8B aligned
      const float* xc = xT + ((size_t)(b0 + m) * TT + t) * TT;  // 8B aligned
      bf16_t* row = atile + m * KP;
#pragma unroll
      for (int h = 0; h < 2; ++h) {  // k = 0..127 via float2 + packed b32
        int k = h * 64 + lane * 2;
        float2 a = *(const float2*)(xr + k);
        float2 b = *(const float2*)(xc + k);
        v2bf pa; pa[0] = (bf16_t)a.x; pa[1] = (bf16_t)a.y;
        v2bf pb; pb[0] = (bf16_t)b.x; pb[1] = (bf16_t)b.y;
        *(v2bf*)(row + k) = pa;
        *(v2bf*)(row + TT + k) = pb;
      }
      if (lane < TT - 128) {  // k = 128..149 tail
        int k = 128 + lane;
        row[k] = (bf16_t)xr[k];
        row[TT + k] = (bf16_t)xc[k];
      }
      if (lane < KP - 2 * TT) row[2 * TT + lane] = (bf16_t)0.f;
      if (it + 1 < TPB) {  // prefetch next timestep's rows (contiguous in t)
        __builtin_prefetch(xr + TT + lane * 5, 0, 3);
        __builtin_prefetch(xc + TT + lane * 5, 0, 3);
      }
    }
    __syncthreads();

    // Bulk-load all 10 A-fragments (one LDS clause), then pure WMMA chain.
    v16bf afr[10];
#pragma unroll
    for (int kb = 0; kb < 10; ++kb) {
      const v8bf* p = (const v8bf*)(atile + mr * KP + kb * 32 + kh * 8);
      afr[kb] = mk16(p[0], p[2]);
    }
    v8f acc;
#pragma unroll
    for (int r = 0; r < 8; ++r) acc[r] = bv;
#pragma unroll
    for (int kb = 0; kb < 10; ++kb)
      acc = __builtin_amdgcn_wmma_f32_16x16x32_bf16(
          false, afr[kb], false, bfr[kb], (short)0, acc, false, false);

    size_t zoff = ((((size_t)t * NBT + bt) * 16 + nt) * 32 + lane) * 8;
    float4* zp = (float4*)(Z + zoff);
    zp[0] = make_float4(acc[0], acc[1], acc[2], acc[3]);
    zp[1] = make_float4(acc[4], acc[5], acc[6], acc[7]);
    __syncthreads();  // protect atile before next staging round
  }
}

// --------------------------- K3: recurrence --------------------------------
// 256 single-wave blocks; block b owns batch tile b for all 150 steps.
__global__ __launch_bounds__(32) void recurrent(
    const float* __restrict__ Z,
    const bf16_t* __restrict__ whbf_g,
    const bf16_t* __restrict__ fcbf,
    const float* __restrict__ thr1p, const float* __restrict__ fc_b,
    const float* __restrict__ betap, const float* __restrict__ thr2p,
    float* __restrict__ out) {
  __shared__ bf16_t wh[GP * HP];    // 32 KB recurrent weights
  __shared__ bf16_t myht[16 * HP];  // 2 KB h repack tile
  int lane = threadIdx.x & 31;

  // Async global->LDS copy of W_hh' (tracked by ASYNCcnt, CDNA5 path).
  {
    unsigned ldsoff = (unsigned)(uintptr_t)(&wh[0]) + lane * 16;
    const char* g = (const char*)whbf_g + lane * 16;
    for (int i = 0; i < (GP * HP * 2) / (32 * 16); ++i) {  // 64 x 512B
      asm volatile("global_load_async_to_lds_b128 %0, %1, off"
                   :
                   : "v"(ldsoff), "v"(g)
                   : "memory");
      ldsoff += 32 * 16;
      g += 32 * 16;
    }
    asm volatile("s_wait_asynccnt 0" ::: "memory");
  }
  __syncthreads();  // single-wave WG: S_NOP, but orders LDS for compiler

  float thr1 = *thr1p, beta = *betap, thr2 = *thr2p;
  int bt = blockIdx.x;
  int nloc = lane & 15, kh = lane >> 4;

  v16bf fcB[2];
#pragma unroll
  for (int kf = 0; kf < 2; ++kf) {
    const v8bf* p = (const v8bf*)(fcbf + nloc * HP + kf * 32 + kh * 8);
    fcB[kf] = mk16(p[0], p[2]);
  }
  float fbias = (nloc < NOUT) ? fc_b[nloc] : 0.f;

  v16bf hfr[2];
#pragma unroll
  for (int kf = 0; kf < 2; ++kf)
#pragma unroll
    for (int e = 0; e < 16; ++e) hfr[kf][e] = (bf16_t)0.f;
  v8f c[4], mem;
#pragma unroll
  for (int jb = 0; jb < 4; ++jb)
#pragma unroll
    for (int r = 0; r < 8; ++r) c[jb][r] = 0.f;
#pragma unroll
  for (int r = 0; r < 8; ++r) mem[r] = 0.f;

  for (int t = 0; t < TT; ++t) {
    const float* zb = Z + (((size_t)t * NBT + bt) * 16) * 256;
    if (t + 1 < TT) {  // prefetch the adjacent 16 KB Z tile (batch tile bt+1) into caches
      const char* pz = (const char*)(zb + 16 * 256);
#pragma unroll
      for (int q = 0; q < 4; ++q)
        __builtin_prefetch(pz + q * 4096 + lane * 128, 0, 3);
    }
#pragma unroll
    for (int jb = 0; jb < 4; ++jb) {
      // Hoist: all 8 W_hh fragments (LDS) + 4 Z accumulators (global) first,
      // then an uninterrupted 8-WMMA chain.
      v16bf bfrs[8];
      v8f g4[4];
#pragma unroll
      for (int gg = 0; gg < 4; ++gg) {
        int nt = jb * 4 + gg;
        const float4* zp = (const float4*)(zb + nt * 256 + lane * 8);
        float4 z0 = zp[0], z1 = zp[1];
        g4[gg][0] = z0.x; g4[gg][1] = z0.y; g4[gg][2] = z0.z; g4[gg][3] = z0.w;
        g4[gg][4] = z1.x; g4[gg][5] = z1.y; g4[gg][6] = z1.z; g4[gg][7] = z1.w;
#pragma unroll
        for (int kf = 0; kf < 2; ++kf) {
          const v8bf* p =
              (const v8bf*)(wh + (nt * 16 + nloc) * HP + kf * 32 + kh * 8);
          bfrs[gg * 2 + kf] = mk16(p[0], p[2]);
        }
      }
#pragma unroll
      for (int gg = 0; gg < 4; ++gg) {
        g4[gg] = __builtin_amdgcn_wmma_f32_16x16x32_bf16(
            false, hfr[0], false, bfrs[gg * 2 + 0], (short)0, g4[gg], false,
            false);
        g4[gg] = __builtin_amdgcn_wmma_f32_16x16x32_bf16(
            false, hfr[1], false, bfrs[gg * 2 + 1], (short)0, g4[gg], false,
            false);
      }
      // lane-local LSTM elementwise (gate permutation aligned i/f/g/o)
#pragma unroll
      for (int r = 0; r < 8; ++r) {
        float si = sigm(g4[0][r]);
        float sf = sigm(g4[1][r]);
        float tg = tanh_fast(g4[2][r]);
        float so = sigm(g4[3][r]);
        float cn = sf * c[jb][r] + si * tg;
        c[jb][r] = cn;
        float h = so * tanh_fast(cn);
        myht[(r + 8 * kh) * HP + jb * 16 + nloc] = (bf16_t)h;  // [m][j]
      }
    }
    __syncthreads();  // free (1-wave WG) ordering fence for LDS repack
#pragma unroll
    for (int kf = 0; kf < 2; ++kf) {
      const v8bf* p = (const v8bf*)(myht + nloc * HP + kf * 32 + kh * 8);
      hfr[kf] = mk16(p[0], p[2]);
    }
    // spike mask as bf16 A-fragments, FC via WMMA
    v16bf s0, s1;
#pragma unroll
    for (int e = 0; e < 16; ++e) {
      s0[e] = (bf16_t)(((float)hfr[0][e] > thr1) ? 1.f : 0.f);
      s1[e] = (bf16_t)(((float)hfr[1][e] > thr1) ? 1.f : 0.f);
    }
    v8f fca;
#pragma unroll
    for (int r = 0; r < 8; ++r) fca[r] = fbias;
    fca = __builtin_amdgcn_wmma_f32_16x16x32_bf16(
        false, s0, false, fcB[0], (short)0, fca, false, false);
    fca = __builtin_amdgcn_wmma_f32_16x16x32_bf16(
        false, s1, false, fcB[1], (short)0, fca, false, false);
    // LIF + outputs
#pragma unroll
    for (int r = 0; r < 8; ++r) {
      float m0 = mem[r];
      float reset = (m0 > thr2) ? 1.f : 0.f;
      float mn = beta * m0 + fca[r] - reset * thr2;
      mem[r] = mn;
      float spk = (mn > thr2) ? 1.f : 0.f;
      if (nloc < NOUT) {
        int b = bt * 16 + r + 8 * kh;
        size_t o = ((size_t)t * BATCH + b) * NOUT + nloc;
        out[o] = spk;
        out[(size_t)TT * BATCH * NOUT + o] = mn;
      }
    }
  }
}

// --------------------------- launch ----------------------------------------
extern "C" void kernel_launch(void* const* d_in, const int* in_sizes, int n_in,
                              void* d_out, int out_size, void* d_ws,
                              size_t ws_size, hipStream_t stream) {
  const float* x    = (const float*)d_in[0];
  const float* w_ih = (const float*)d_in[1];
  const float* w_hh = (const float*)d_in[2];
  const float* b_ih = (const float*)d_in[3];
  const float* b_hh = (const float*)d_in[4];
  const float* thr1 = (const float*)d_in[5];
  const float* fc_w = (const float*)d_in[6];
  const float* fc_b = (const float*)d_in[7];
  const float* beta = (const float*)d_in[8];
  const float* thr2 = (const float*)d_in[9];
  float* out = (float*)d_out;

  // Workspace partition (~1.0 GB total), 256B-aligned sections.
  char* ws = (char*)d_ws;
  size_t off = 0;
  auto carve = [&](size_t bytes) {
    char* p = ws + off;
    off = (off + bytes + 255) & ~(size_t)255;
    return (void*)p;
  };
  float*  xT   = (float*) carve((size_t)BATCH * TT * TT * 4);  // 369 MB
  bf16_t* wbf  = (bf16_t*)carve((size_t)GP * KP * 2);
  bf16_t* whbf = (bf16_t*)carve((size_t)GP * HP * 2);
  float*  bias = (float*) carve((size_t)GP * 4);
  bf16_t* fcbf = (bf16_t*)carve((size_t)16 * HP * 2);
  float*  Zbuf = (float*) carve((size_t)TT * NBT * 16 * 256 * 4);  // 630 MB

  prep_weights<<<1, 256, 0, stream>>>(w_ih, w_hh, b_ih, b_hh, fc_w, wbf, whbf,
                                      bias, fcbf);
  transpose_x<<<dim3(100, BATCH), 256, 0, stream>>>(x, xT);
  input_gemm<<<NBT * (TT / TPB), 512, 0, stream>>>(x, xT, wbf, bias, Zbuf);
  recurrent<<<NBT, 32, 0, stream>>>(Zbuf, whbf, fcbf, thr1, fc_b, beta, thr2,
                                    out);
}
